// TransformerEncoderLayer_38671885533239
// MI455X (gfx1250) — matrix operand order repaired and checked by reading, not compiled, not run on hardware
//
#include <hip/hip_runtime.h>
#include <hip/hip_bf16.h>

// ---------------------------------------------------------------------------
// Types & helpers
// ---------------------------------------------------------------------------
typedef unsigned short u16;
typedef __attribute__((ext_vector_type(16))) __bf16 bf16x16;
typedef __attribute__((ext_vector_type(8)))  float  f32x8;
typedef __attribute__((ext_vector_type(4)))  unsigned int u32x4;
typedef __attribute__((ext_vector_type(8)))  int i32x8;
typedef __attribute__((ext_vector_type(4)))  int i32x4;

union FragU { uint4 q[2]; bf16x16 v; };
union U4U   { uint4 q; u16 s[8]; };

__device__ __forceinline__ u16 f2bf(float f) {
    unsigned u = __float_as_uint(f);
    u += 0x7FFFu + ((u >> 16) & 1u);      // round-to-nearest-even
    return (u16)(u >> 16);
}
__device__ __forceinline__ float bf2f(u16 h) {
    return __uint_as_float(((unsigned)h) << 16);
}

// Load a 32-byte WMMA fragment from two 16-byte runs (p1 = p0+8 for contiguous).
__device__ __forceinline__ bf16x16 load_frag(const u16* p0, const u16* p1) {
    FragU f;
    f.q[0] = *(const uint4*)p0;
    f.q[1] = *(const uint4*)p1;
    return f.v;
}

__device__ __forceinline__ f32x8 wmma_bf16(bf16x16 a, bf16x16 b, f32x8 c) {
    return __builtin_amdgcn_wmma_f32_16x16x32_bf16(
        false, a, false, b, (short)0, c, false, false);
}

// ---------------------------------------------------------------------------
// TDM: 2D bf16 tile load Global -> LDS (Tensor Data Mover).
//   tile_w elements per row (bf16), tile_h rows, global row stride (elements).
//   LDS padding via TDM pad: 4 DWORDs (8 bf16) after every 16 DWORDs (32 bf16)
//   -> LDS row stride of 40 u16, matching the [rows][40] shared arrays.
//   Descriptor bitfields per CDNA5 ISA (D# groups 0/1).
//   This toolchain declares the 6-arg builtin:
//   (u32x4 g0, i32x8 g1, i32x4, i32x4, i32x8, i32 cpol).
// ---------------------------------------------------------------------------
__device__ __forceinline__ void tdm_load_2d(unsigned lds_off, const u16* gaddr,
                                            unsigned tile_w, unsigned tile_h,
                                            unsigned row_stride,
                                            unsigned tensor_w, unsigned tensor_h) {
    unsigned long long ga = (unsigned long long)(uintptr_t)gaddr;
    u32x4 g0;
    g0[0] = 1u;                                       // count=1 (valid user D#)
    g0[1] = lds_off;                                  // lds_addr [63:32]
    g0[2] = (unsigned)(ga & 0xFFFFFFFFu);             // global_addr lo
    g0[3] = (unsigned)((ga >> 32) & 0x01FFFFFFu)      // global_addr hi (bits 56:32)
          | (2u << 30);                               // type=2 ("image")

    const unsigned data_size    = 1;  // 2-byte elements
    const unsigned pad_enable   = 1;
    const unsigned pad_interval = 3;  // 16 DWORDs before padding (= 32 bf16)
    const unsigned pad_amount   = 3;  // 4 DWORDs of padding (= 8 bf16)
    i32x8 g1;
    g1[0] = (int)((data_size << 16) | (pad_enable << 20) |
                  (pad_interval << 22) | (pad_amount << 25));
    g1[1] = (int)((tensor_w & 0xFFFFu) << 16);                        // dim0 lo16
    g1[2] = (int)((tensor_w >> 16) | ((tensor_h & 0xFFFFu) << 16));   // dim0 hi / dim1 lo
    g1[3] = (int)((tensor_h >> 16) | (tile_w << 16));                 // dim1 hi / tile_dim0
    g1[4] = (int)(tile_h & 0xFFFFu);                                  // tile_dim1 (tile_dim2=0)
    g1[5] = (int)row_stride;                                          // dim0_stride lo32
    g1[6] = 0;                                                        // dim0_stride hi / dim1_stride lo
    g1[7] = 0;

    i32x4 z4 = {};
    i32x8 z8 = {};
    __builtin_amdgcn_tensor_load_to_lds(g0, g1, z4, z4, z8, 0);
}

// LDS byte offset of a __shared__ object (generic addr low 32 bits map to LDS).
__device__ __forceinline__ unsigned lds_offset(const void* p) {
    return (unsigned)(uintptr_t)p;
}

// ---------------------------------------------------------------------------
// Problem constants
// ---------------------------------------------------------------------------
#define BATCH 4
#define SEQ   1024
#define DEP   256
#define HEADS 16
#define FFD   128
#define NTOK  (BATCH * SEQ)     // 4096
#define DH    (DEP * HEADS)     // 4096

// ---------------------------------------------------------------------------
// Kernel: f32 -> bf16 cast (plain) and cast+transpose (for weights: B -> B^T)
// ---------------------------------------------------------------------------
__global__ void cast_f32_bf16(const float* __restrict__ in,
                              u16* __restrict__ out, int n) {
    int i = blockIdx.x * blockDim.x + threadIdx.x;
    if (i < n) out[i] = f2bf(in[i]);
}

// in: R x C row-major f32  ->  out: C x R row-major bf16 (transposed)
__global__ void cast_transpose_f32_bf16(const float* __restrict__ in,
                                        u16* __restrict__ out, int R, int C) {
    int i = blockIdx.x * blockDim.x + threadIdx.x;
    if (i < R * C) {
        int r = i / C, c = i % C;
        out[(size_t)c * R + r] = f2bf(in[i]);
    }
}

// ---------------------------------------------------------------------------
// Kernel: bf16 GEMM (TN form), f32 accumulation via WMMA, TDM tile staging.
//   C[M,N] = A[M,K] @ B[K,N] with B supplied pre-transposed: BT[N,K] row-major.
//   Block = 256 threads (8 waves), tile 128x64, K-step 32.
//   Wave w owns rows w*16..+15 and all 64 columns: 1 A-frag reused by 4 WMMAs.
//   Tiles staged Global->LDS by wave 0 via tensor_load_to_lds (TENSORcnt).
//   Epilogue: optional f32 bias, optional ReLU, output f32 or bf16.
//   Requires M%128==0, N%64==0, K%32==0 (true for all uses here).
// ---------------------------------------------------------------------------
__global__ void __launch_bounds__(256)
gemm_bf16_tn(const u16* __restrict__ A, const u16* __restrict__ BT,
             int M, int N, int K,
             const float* __restrict__ bias, int relu,
             float* __restrict__ Cf, u16* __restrict__ Cb) {
    __shared__ __attribute__((aligned(16))) u16 As[128][40]; // 128 rows x 32 k
    __shared__ __attribute__((aligned(16))) u16 Bs[64][40];  // 64 cols x 32 k

    const int tid  = threadIdx.x;
    const int lane = tid & 31;
    const int wave = tid >> 5;
    const int m0   = blockIdx.y * 128;
    const int n0   = blockIdx.x * 64;

    f32x8 c[4] = {{}, {}, {}, {}};

    const unsigned asOff = lds_offset(&As[0][0]);
    const unsigned bsOff = lds_offset(&Bs[0][0]);

    for (int k0 = 0; k0 < K; k0 += 32) {
        if (wave == 0) {
            tdm_load_2d(asOff, &A [(size_t)m0 * K + k0], 32, 128, (unsigned)K,
                        (unsigned)K, (unsigned)M);
            tdm_load_2d(bsOff, &BT[(size_t)n0 * K + k0], 32, 64, (unsigned)K,
                        (unsigned)K, (unsigned)N);
            __builtin_amdgcn_s_wait_tensorcnt(0);
        }
        __syncthreads();

        // A fragment (16x32): lane row = wave*16 + lane%16;
        // low lanes k=0..15, high lanes k=16..31.
        const int ar = wave * 16 + (lane & 15);
        const int ak = (lane < 16) ? 0 : 16;
        bf16x16 af = load_frag(&As[ar][ak], &As[ar][ak + 8]);

        // B fragments (32x16): lane col = t*16 + lane%16;
        // low lanes k=0..15, high lanes k=16..31 (contiguous rows of BT).
        const int bk = (lane < 16) ? 0 : 16;
        #pragma unroll
        for (int t = 0; t < 4; ++t) {
            const int bc = t * 16 + (lane & 15);
            bf16x16 bf = load_frag(&Bs[bc][bk], &Bs[bc][bk + 8]);
            c[t] = wmma_bf16(af, bf, c[t]);
        }
        __syncthreads();
    }

    // Epilogue. C layout: VGPR r -> row r (lanes 0-15) / r+8 (lanes 16-31), col = lane%16.
    const int lrow = (lane < 16) ? 0 : 8;
    const int lcol = lane & 15;
    #pragma unroll
    for (int t = 0; t < 4; ++t) {
        #pragma unroll
        for (int r = 0; r < 8; ++r) {
            int grow = m0 + wave * 16 + lrow + r;
            int gcol = n0 + t * 16 + lcol;
            float v = c[t][r];
            if (bias) v += bias[gcol];
            if (relu) v = fmaxf(v, 0.0f);
            if (Cb) Cb[(size_t)grow * N + gcol] = f2bf(v);
            else    Cf[(size_t)grow * N + gcol] = v;
        }
    }
}

// ---------------------------------------------------------------------------
// Kernel: fused attention for one (b, h, 16-query tile).
//   Q,K,V bf16 laid out as [b, s, h, d] (d fastest, stride H*D between s).
//   Phase 1: scores (16x1024) via WMMA -> bf16 in LDS (scaled by 1/sqrt(D)).
//   Softmax in f32 (shuffle reductions), P kept unnormalized, 1/l saved.
//   Phase 2: P @ V with WMMA over 32-key slabs staged transposed in LDS.
//   Output heads[tok, h*D + d] bf16, normalized by 1/l.
// ---------------------------------------------------------------------------
__global__ void __launch_bounds__(256)
attn_fused(const u16* __restrict__ Q, const u16* __restrict__ K,
           const u16* __restrict__ V, u16* __restrict__ Hout) {
    __shared__ __attribute__((aligned(16))) u16 sc[16][1024];  // scores/P  32 KB
    __shared__ __attribute__((aligned(16))) u16 vtT[256][40];  // V^T slab  20 KB
    __shared__ float linv[16];

    const int qt = blockIdx.x & 63;
    const int h  = (blockIdx.x >> 6) & 15;
    const int b  = blockIdx.x >> 10;
    const int q0 = qt * 16;

    const int tid  = threadIdx.x;
    const int lane = tid & 31;
    const int wave = tid >> 5;

    // ---- Phase 1: scores -------------------------------------------------
    const u16* qrow = Q + (((size_t)b * SEQ + q0 + (lane & 15)) * HEADS + h) * DEP;
    const int  aoff = (lane < 16) ? 0 : 16;
    bf16x16 qa[8];
    #pragma unroll
    for (int f = 0; f < 8; ++f)
        qa[f] = load_frag(qrow + f * 32 + aoff, qrow + f * 32 + aoff + 8);

    const int bk = (lane < 16) ? 0 : 16;
    for (int j = wave; j < 64; j += 8) {           // 8 key tiles per wave
        const u16* krow = K + (((size_t)b * SEQ + j * 16 + (lane & 15)) * HEADS + h) * DEP;
        if (j + 8 < 64) {                           // prefetch next key tile rows
            __builtin_prefetch(krow + (size_t)8 * 16 * HEADS * DEP, 0, 1);
        }
        f32x8 acc = {};
        #pragma unroll
        for (int f = 0; f < 8; ++f) {
            bf16x16 kb = load_frag(krow + f * 32 + bk, krow + f * 32 + bk + 8);
            acc = wmma_bf16(qa[f], kb, acc);
        }
        const int lrow = (lane < 16) ? 0 : 8;
        const int col  = j * 16 + (lane & 15);
        #pragma unroll
        for (int r = 0; r < 8; ++r)
            sc[lrow + r][col] = f2bf(acc[r] * 0.0625f);   // 1/sqrt(256)
    }
    __syncthreads();

    // ---- Softmax over 1024 keys per row (16 threads per row) -------------
    {
        const int row = tid >> 4;
        const int i   = tid & 15;
        const int cbeg = i * 64, cend = cbeg + 64;
        float m = -3.4e38f;
        for (int c = cbeg; c < cend; ++c) m = fmaxf(m, bf2f(sc[row][c]));
        #pragma unroll
        for (int off = 8; off > 0; off >>= 1) m = fmaxf(m, __shfl_xor(m, off));
        float s = 0.0f;
        for (int c = cbeg; c < cend; ++c) {
            float p = __expf(bf2f(sc[row][c]) - m);
            s += p;
            sc[row][c] = f2bf(p);
        }
        #pragma unroll
        for (int off = 8; off > 0; off >>= 1) s += __shfl_xor(s, off);
        if (i == 0) linv[row] = 1.0f / s;
    }
    __syncthreads();

    // ---- Phase 2: P @ V --------------------------------------------------
    f32x8 oc0 = {};
    f32x8 oc1 = {};
    const int nt0 = wave * 16;           // this wave's output d-columns
    const int nt1 = (wave + 8) * 16;

    for (int c = 0; c < 32; ++c) {       // 32-key slabs
        // Stage V slab transposed: vtT[d][k], 8192 u16 total.
        #pragma unroll
        for (int it = 0; it < 4; ++it) {
            int e   = it * 2048 + tid * 8;
            int key = e >> 8;
            int d0  = e & 255;
            U4U vv;
            vv.q = *(const uint4*)&V[(((size_t)b * SEQ + c * 32 + key) * HEADS + h) * DEP + d0];
            #pragma unroll
            for (int ii = 0; ii < 8; ++ii) vtT[d0 + ii][key] = vv.s[ii];
        }
        __syncthreads();

        // P A-fragment for keys c*32..+31.
        const int prow = lane & 15;
        const int pk   = c * 32 + ((lane < 16) ? 0 : 16);
        bf16x16 pf = load_frag(&sc[prow][pk], &sc[prow][pk + 8]);

        // V B-fragments.
        const int vc0 = nt0 + (lane & 15);
        const int vc1 = nt1 + (lane & 15);
        bf16x16 vf0 = load_frag(&vtT[vc0][bk], &vtT[vc0][bk + 8]);
        bf16x16 vf1 = load_frag(&vtT[vc1][bk], &vtT[vc1][bk + 8]);
        oc0 = wmma_bf16(pf, vf0, oc0);
        oc1 = wmma_bf16(pf, vf1, oc1);
        __syncthreads();
    }

    // ---- Epilogue: normalize & store heads -------------------------------
    const int lrow = (lane < 16) ? 0 : 8;
    const int lcol = lane & 15;
    #pragma unroll
    for (int r = 0; r < 8; ++r) {
        int rl  = lrow + r;
        size_t tok = (size_t)b * SEQ + q0 + rl;
        float sc1 = linv[rl];
        Hout[tok * DH + h * DEP + nt0 + lcol] = f2bf(oc0[r] * sc1);
        Hout[tok * DH + h * DEP + nt1 + lcol] = f2bf(oc1[r] * sc1);
    }
}

// ---------------------------------------------------------------------------
// Kernel: y = LayerNorm(a + r) * g + be ; optional bf16 copy of y.
//   One block per token row, 256 threads == DEP elements.
// ---------------------------------------------------------------------------
__global__ void __launch_bounds__(256)
add_layernorm(const float* __restrict__ a, const float* __restrict__ r,
              const float* __restrict__ g, const float* __restrict__ be,
              float* __restrict__ y, u16* __restrict__ ybf) {
    __shared__ float red[256];
    const int row = blockIdx.x;
    const int t   = threadIdx.x;
    const size_t idx = (size_t)row * DEP + t;

    float v = a[idx] + r[idx];
    red[t] = v;
    __syncthreads();
    #pragma unroll
    for (int o = 128; o > 0; o >>= 1) {
        if (t < o) red[t] += red[t + o];
        __syncthreads();
    }
    float mu = red[0] * (1.0f / DEP);
    __syncthreads();

    float dv = v - mu;
    red[t] = dv * dv;
    __syncthreads();
    #pragma unroll
    for (int o = 128; o > 0; o >>= 1) {
        if (t < o) red[t] += red[t + o];
        __syncthreads();
    }
    float var = red[0] * (1.0f / DEP);
    float rs  = rsqrtf(var + 1e-5f);

    float out = dv * rs * g[t] + be[t];
    y[idx] = out;
    if (ybf) ybf[idx] = f2bf(out);
}

// ---------------------------------------------------------------------------
// Host launcher
// ---------------------------------------------------------------------------
extern "C" void kernel_launch(void* const* d_in, const int* in_sizes, int n_in,
                              void* d_out, int out_size, void* d_ws, size_t ws_size,
                              hipStream_t stream) {
    (void)in_sizes; (void)n_in; (void)out_size; (void)ws_size;

    const float* seq   = (const float*)d_in[0];
    const float* wq    = (const float*)d_in[1];
    const float* wk    = (const float*)d_in[2];
    const float* wv    = (const float*)d_in[3];
    const float* wo    = (const float*)d_in[4];
    const float* ln1_g = (const float*)d_in[5];
    const float* ln1_b = (const float*)d_in[6];
    const float* w1    = (const float*)d_in[7];
    const float* b1    = (const float*)d_in[8];
    const float* w2    = (const float*)d_in[9];
    const float* b2    = (const float*)d_in[10];
    const float* ln2_g = (const float*)d_in[11];
    const float* ln2_b = (const float*)d_in[12];

    char* wsp = (char*)d_ws;
    auto alloc = [&](size_t bytes) -> char* {
        char* p = wsp;
        wsp += (bytes + 255) & ~(size_t)255;
        return p;
    };

    u16*   seqb = (u16*)  alloc((size_t)NTOK * DEP * 2);
    u16*   wqT  = (u16*)  alloc((size_t)DEP * DH * 2);   // [DH][DEP]
    u16*   wkT  = (u16*)  alloc((size_t)DEP * DH * 2);
    u16*   wvT  = (u16*)  alloc((size_t)DEP * DH * 2);
    u16*   woT  = (u16*)  alloc((size_t)DH * DEP * 2);   // [DEP][DH]
    u16*   w1T  = (u16*)  alloc((size_t)DEP * FFD * 2);  // [FFD][DEP]
    u16*   w2T  = (u16*)  alloc((size_t)FFD * DEP * 2);  // [DEP][FFD]
    u16*   Qb   = (u16*)  alloc((size_t)NTOK * DH * 2);
    u16*   Kb   = (u16*)  alloc((size_t)NTOK * DH * 2);
    u16*   Vb   = (u16*)  alloc((size_t)NTOK * DH * 2);
    u16*   Hb   = (u16*)  alloc((size_t)NTOK * DH * 2);
    float* Of   = (float*)alloc((size_t)NTOK * DEP * 4);
    float* Xf   = (float*)alloc((size_t)NTOK * DEP * 4);
    u16*   Xb   = (u16*)  alloc((size_t)NTOK * DEP * 2);
    u16*   H1   = (u16*)  alloc((size_t)NTOK * FFD * 2);
    float* Ff   = (float*)alloc((size_t)NTOK * DEP * 4);

    cast_f32_bf16<<<(NTOK * DEP + 255) / 256, 256, 0, stream>>>(seq, seqb, NTOK * DEP);
    auto castT = [&](const float* src, u16* dst, int R, int C) {
        cast_transpose_f32_bf16<<<(R * C + 255) / 256, 256, 0, stream>>>(src, dst, R, C);
    };
    castT(wq, wqT, DEP, DH);    // wq: [DEP,DH] -> wqT: [DH,DEP]
    castT(wk, wkT, DEP, DH);
    castT(wv, wvT, DEP, DH);
    castT(wo, woT, DH, DEP);    // wo: [DH,DEP] -> woT: [DEP,DH]
    castT(w1, w1T, DEP, FFD);   // w1: [DEP,FFD] -> w1T: [FFD,DEP]
    castT(w2, w2T, FFD, DEP);   // w2: [FFD,DEP] -> w2T: [DEP,FFD]

    // QKV projections: [4096,256] @ [256,4096] -> bf16 [b,s,h,d]
    gemm_bf16_tn<<<dim3(DH / 64, NTOK / 128), 256, 0, stream>>>(
        seqb, wqT, NTOK, DH, DEP, nullptr, 0, nullptr, Qb);
    gemm_bf16_tn<<<dim3(DH / 64, NTOK / 128), 256, 0, stream>>>(
        seqb, wkT, NTOK, DH, DEP, nullptr, 0, nullptr, Kb);
    gemm_bf16_tn<<<dim3(DH / 64, NTOK / 128), 256, 0, stream>>>(
        seqb, wvT, NTOK, DH, DEP, nullptr, 0, nullptr, Vb);

    // Fused attention: one block per (b, h, 16-query tile).
    attn_fused<<<BATCH * HEADS * (SEQ / 16), 256, 0, stream>>>(Qb, Kb, Vb, Hb);

    // Output projection: [4096,4096] @ [4096,256] -> f32
    gemm_bf16_tn<<<dim3(DEP / 64, NTOK / 128), 256, 0, stream>>>(
        Hb, woT, NTOK, DEP, DH, nullptr, 0, Of, nullptr);

    // x = LN(seq + o)
    add_layernorm<<<NTOK, 256, 0, stream>>>(seq, Of, ln1_g, ln1_b, Xf, Xb);

    // FFN: relu(x @ w1 + b1) @ w2 + b2
    gemm_bf16_tn<<<dim3(FFD / 64, NTOK / 128), 256, 0, stream>>>(
        Xb, w1T, NTOK, FFD, DEP, b1, 1, nullptr, H1);
    gemm_bf16_tn<<<dim3(DEP / 64, NTOK / 128), 256, 0, stream>>>(
        H1, w2T, NTOK, DEP, FFD, b2, 0, Ff, nullptr);

    // out = LN(x + ff)
    add_layernorm<<<NTOK, 256, 0, stream>>>(Xf, Ff, ln2_g, ln2_b,
                                            (float*)d_out, nullptr);
}
